// MultiHeadAttention_43834436223245
// MI455X (gfx1250) — compile-verified, matrix operand order checked by reading, not run on hardware
//
#include <hip/hip_runtime.h>

// ---------------------------------------------------------------------------
// Problem constants (from reference): B=2, S=2048, DIN=DOUT=2048, H=16, D=128
// ---------------------------------------------------------------------------
#define SEQ   2048
#define DIN   2048
#define NQKV  6144
#define MTOT  4096          // B*S
#define NHEAD 16
#define HDIM  128

typedef __attribute__((ext_vector_type(16))) __bf16 v16bf;
typedef __attribute__((ext_vector_type(8)))  float  v8f;

// GCC-style int4 vector — matches the async-to-LDS builtin's pointee type
typedef int gv4i __attribute__((vector_size(16)));
typedef __attribute__((address_space(1))) gv4i gv4i_as1;
typedef __attribute__((address_space(3))) gv4i gv4i_as3;

#ifndef __has_builtin
#define __has_builtin(x) 0
#endif
#if __has_builtin(__builtin_amdgcn_global_load_async_to_lds_b128)
#define HAVE_ASYNC_LDS 1
#else
#define HAVE_ASYNC_LDS 0
#endif

#if __has_builtin(__builtin_amdgcn_s_wait_asynccnt)
#define WAIT_ASYNC(n) __builtin_amdgcn_s_wait_asynccnt(n)
#else
#define WAIT_ASYNC(n) asm volatile("s_wait_asynccnt %0" ::"i"(n) : "memory")
#endif

union ABFrag {
  uint4          u[2];   // two b128 loads
  unsigned short h[16];  // 16 bf16 elements
  v16bf          v;      // WMMA operand
};

__device__ __forceinline__ unsigned short f2bf(float f) {
  unsigned u = __float_as_uint(f);
  u += 0x7FFFu + ((u >> 16) & 1u);          // round-to-nearest-even
  return (unsigned short)(u >> 16);
}

__device__ __forceinline__ v8f v8f_zero() {
  v8f r = {0.f, 0.f, 0.f, 0.f, 0.f, 0.f, 0.f, 0.f};
  return r;
}

__device__ __forceinline__ v8f wmma_bf16(const v16bf& a, const v16bf& b, v8f c) {
  return __builtin_amdgcn_wmma_f32_16x16x32_bf16(false, a, false, b, (short)0, c,
                                                 false, false);
}

// ---------------------------------------------------------------------------
// f32 -> bf16 straight convert
// ---------------------------------------------------------------------------
__global__ void __launch_bounds__(256)
cvt_bf16_kernel(const float* __restrict__ in, unsigned short* __restrict__ out,
                int n) {
  int i = blockIdx.x * 256 + threadIdx.x;
  if (i < n) out[i] = f2bf(in[i]);
}

// ---------------------------------------------------------------------------
// Transpose + convert: in[K][N] f32 -> out[N][K] bf16  (coalesced writes)
// ---------------------------------------------------------------------------
__global__ void __launch_bounds__(256)
transpose_bf16_kernel(const float* __restrict__ in, unsigned short* __restrict__ out,
                      int K, int N) {
  long long i = (long long)blockIdx.x * 256 + threadIdx.x;
  if (i >= (long long)K * N) return;
  int k = (int)(i % K);
  int n = (int)(i / K);
  out[(size_t)n * K + k] = f2bf(in[(size_t)k * N + n]);
}

// ---------------------------------------------------------------------------
// Shared double-buffered GEMM core.
//   A  [.][2048] bf16 row-major, Bt [.][2048] bf16 (B column-major)
//   Block: 256 threads = 8 waves (4 M-waves x 2 N-waves), tile 128x128.
//   Each wave: 32x64 -> 8 accumulators; K-step 32 staged via LDS.
//   LDS rows padded to 40 bf16 (80 B = 5x16 B: keeps b128 alignment,
//   conflict-free row stride).
// Stage fill uses GLOBAL_LOAD_ASYNC_TO_LDS_B128 (ASYNCcnt) when available.
// ---------------------------------------------------------------------------
#define LROW 40

__device__ __forceinline__ void stage_tiles(
    const unsigned short* __restrict__ A, const unsigned short* __restrict__ Bt,
    int row0b, int col0b, int kk,
    unsigned short (&Al)[2][128][LROW], unsigned short (&Bl)[2][128][LROW],
    int buf) {
  const int tid = threadIdx.x;
#pragma unroll
  for (int half = 0; half < 2; ++half) {
    const int row = (tid >> 2) + half * 64;
    const int ch  = (tid & 3) * 8;
    const unsigned short* gA = A  + (size_t)(row0b + row) * DIN + kk + ch;
    const unsigned short* gB = Bt + (size_t)(col0b + row) * DIN + kk + ch;
    unsigned short* lA = &Al[buf][row][ch];
    unsigned short* lB = &Bl[buf][row][ch];
#if HAVE_ASYNC_LDS
    __builtin_amdgcn_global_load_async_to_lds_b128((gv4i_as1*)gA, (gv4i_as3*)lA,
                                                   0, 0);
    __builtin_amdgcn_global_load_async_to_lds_b128((gv4i_as1*)gB, (gv4i_as3*)lB,
                                                   0, 0);
#else
    *(uint4*)lA = *(const uint4*)gA;
    *(uint4*)lB = *(const uint4*)gB;
#endif
  }
}

__device__ __forceinline__ void gemm_core_128x128(
    const unsigned short* __restrict__ A, const unsigned short* __restrict__ Bt,
    int row0b, int col0b,
    unsigned short (&Al)[2][128][LROW], unsigned short (&Bl)[2][128][LROW],
    v8f (&acc)[2][4]) {
  const int lane = threadIdx.x & 31;
  const int wave = threadIdx.x >> 5;
  const int mrow = lane & 15;
  const int hi   = lane >> 4;
  const int wm   = wave & 3;
  const int wn   = wave >> 2;

#pragma unroll
  for (int tm = 0; tm < 2; ++tm)
#pragma unroll
    for (int tn = 0; tn < 4; ++tn) acc[tm][tn] = v8f_zero();

  const int NSTEP = DIN / 32;
  stage_tiles(A, Bt, row0b, col0b, 0, Al, Bl, 0);

  for (int s = 0; s < NSTEP; ++s) {
    const int buf = s & 1;
    if (s + 1 < NSTEP) {
      stage_tiles(A, Bt, row0b, col0b, (s + 1) * 32, Al, Bl, buf ^ 1);
#if HAVE_ASYNC_LDS
      WAIT_ASYNC(4);                     // retire stage s, keep s+1 in flight
#endif
    } else {
#if HAVE_ASYNC_LDS
      WAIT_ASYNC(0);
#endif
    }
    __syncthreads();

    ABFrag af[2];
#pragma unroll
    for (int tm = 0; tm < 2; ++tm) {
      const int r = wm * 32 + tm * 16 + mrow;
      af[tm].u[0] = *(const uint4*)&Al[buf][r][hi * 8];
      af[tm].u[1] = *(const uint4*)&Al[buf][r][16 + hi * 8];
    }
#pragma unroll
    for (int tn = 0; tn < 4; ++tn) {
      const int c = wn * 64 + tn * 16 + mrow;
      ABFrag bfr;
      bfr.u[0] = *(const uint4*)&Bl[buf][c][hi * 8];
      bfr.u[1] = *(const uint4*)&Bl[buf][c][16 + hi * 8];
      acc[0][tn] = wmma_bf16(af[0].v, bfr.v, acc[0][tn]);
      acc[1][tn] = wmma_bf16(af[1].v, bfr.v, acc[1][tn]);
    }
    __syncthreads();
  }
}

// ---------------------------------------------------------------------------
// QKV GEMM with fused RoPE epilogue.
//   qb : [B*H][S][128] bf16 (RoPE + 1/sqrt(128) folded in)
//   kb : [B*H][S][128] bf16 (RoPE)
//   vt : [B*H][128][S] bf16 (transposed for P@V B-operand)
// ---------------------------------------------------------------------------
__global__ void __launch_bounds__(256)
gemm_qkv_rope_kernel(const unsigned short* __restrict__ A,
                     const unsigned short* __restrict__ Bt,
                     unsigned short* __restrict__ qb,
                     unsigned short* __restrict__ kb,
                     unsigned short* __restrict__ vt) {
  __shared__ unsigned short Al[2][128][LROW];
  __shared__ unsigned short Bl[2][128][LROW];
  const int lane = threadIdx.x & 31;
  const int wave = threadIdx.x >> 5;
  const int mrow = lane & 15;
  const int hi   = lane >> 4;
  const int wm   = wave & 3;
  const int wn   = wave >> 2;
  const int row0b = blockIdx.y * 128;
  const int col0b = blockIdx.x * 128;

  v8f acc[2][4];
  gemm_core_128x128(A, Bt, row0b, col0b, Al, Bl, acc);

  const int col0 = col0b + wn * 64;
  const float qscale = 0.08838834764831845f;        // 1/sqrt(128)
  const float l2t_64 = 0.20762050593046f;           // log2(10000)/64

#pragma unroll
  for (int tm = 0; tm < 2; ++tm) {
    const int rowt  = row0b + wm * 32 + tm * 16;
    const int batch = rowt >> 11;
    const int sbase = rowt & (SEQ - 1);
#pragma unroll
    for (int tn = 0; tn < 4; ++tn) {
      const int n = col0 + tn * 16 + mrow;
      const int region = n >> 11;          // 0=q 1=k 2=v (wave-uniform)
      const int nr = n & 2047;
      const int h = nr >> 7;
      const int d = nr & 127;
      if (region == 2) {
        // V transposed: vt[(b*16+h)][d][s], 8 consecutive s -> one b128 store
        unsigned short hv[8];
#pragma unroll
        for (int j = 0; j < 8; ++j) hv[j] = f2bf(acc[tm][tn][j]);
        uint4 st;
        st.x = (unsigned)hv[0] | ((unsigned)hv[1] << 16);
        st.y = (unsigned)hv[2] | ((unsigned)hv[3] << 16);
        st.z = (unsigned)hv[4] | ((unsigned)hv[5] << 16);
        st.w = (unsigned)hv[6] | ((unsigned)hv[7] << 16);
        size_t base =
            ((size_t)(batch * NHEAD + h) * HDIM + d) * SEQ + sbase + hi * 8;
        *(uint4*)(vt + base) = st;
      } else {
        unsigned short* dst = (region == 0) ? qb : kb;
        const float inv_freq = exp2f(-(float)(d >> 1) * l2t_64);
#pragma unroll
        for (int j = 0; j < 8; ++j) {
          const int srow = sbase + j + hi * 8;
          float val = acc[tm][tn][j];
          float partner = __shfl_xor(val, 1, 32);   // paired rotary dim
          float ang = (float)srow * inv_freq;
          float sn = __sinf(ang);
          float cs = __cosf(ang);
          float o = (d & 1) ? (partner * sn + val * cs)
                            : (val * cs - partner * sn);
          if (region == 0) o *= qscale;
          dst[((size_t)(batch * NHEAD + h) * SEQ + srow) * HDIM + d] = f2bf(o);
        }
      }
    }
  }
}

// ---------------------------------------------------------------------------
// Output projection GEMM: out[MTOT][2048] f32
// ---------------------------------------------------------------------------
__global__ void __launch_bounds__(256)
gemm_proj_kernel(const unsigned short* __restrict__ A,    // yb [MTOT][DIN]
                 const unsigned short* __restrict__ Bt,   // WprojT [2048][DIN]
                 float* __restrict__ Cout) {
  __shared__ unsigned short Al[2][128][LROW];
  __shared__ unsigned short Bl[2][128][LROW];
  const int lane = threadIdx.x & 31;
  const int wave = threadIdx.x >> 5;
  const int mrow = lane & 15;
  const int hi   = lane >> 4;
  const int wm   = wave & 3;
  const int wn   = wave >> 2;
  const int row0b = blockIdx.y * 128;
  const int col0b = blockIdx.x * 128;

  v8f acc[2][4];
  gemm_core_128x128(A, Bt, row0b, col0b, Al, Bl, acc);

#pragma unroll
  for (int tm = 0; tm < 2; ++tm) {
    const int rowt = row0b + wm * 32 + tm * 16;
#pragma unroll
    for (int tn = 0; tn < 4; ++tn) {
      const int col = col0b + wn * 64 + tn * 16 + mrow;
#pragma unroll
      for (int j = 0; j < 8; ++j) {
        Cout[(size_t)(rowt + j + hi * 8) * DIN + col] = acc[tm][tn][j];
      }
    }
  }
}

// ---------------------------------------------------------------------------
// Flash attention forward. One wave per 16-query tile; 32-key steps.
//   qb/kb : [B*H][S][128] bf16,  vt : [B*H][128][S] bf16
//   yb    : [B][S][H*128] bf16
// ---------------------------------------------------------------------------
__global__ void __launch_bounds__(128)
attn_fwd_kernel(const unsigned short* __restrict__ qb,
                const unsigned short* __restrict__ kb,
                const unsigned short* __restrict__ vt,
                unsigned short* __restrict__ yb) {
  __shared__ float Pbuf[4][16 * 32];
  const int lane = threadIdx.x & 31;
  const int wave = threadIdx.x >> 5;
  const int mrow = lane & 15;
  const int hi   = lane >> 4;
  const int bh   = blockIdx.y;                      // 0..31
  const int q0   = blockIdx.x * 64 + wave * 16;
  const unsigned short* qh = qb + (size_t)bh * SEQ * HDIM;
  const unsigned short* kh = kb + (size_t)bh * SEQ * HDIM;
  const unsigned short* vh = vt + (size_t)bh * HDIM * SEQ;
  float* P = Pbuf[wave];

  // Q fragments: 16 queries x 128 dims = 4 x (16x32) A-operands
  ABFrag Aq[4];
#pragma unroll
  for (int c = 0; c < 4; ++c) {
    const uint4* ap =
        (const uint4*)(qh + (size_t)(q0 + mrow) * HDIM + c * 32 + hi * 8);
    Aq[c].u[0] = ap[0];
    Aq[c].u[1] = ap[2];
  }

  v8f O[8] = {v8f_zero(), v8f_zero(), v8f_zero(), v8f_zero(),
              v8f_zero(), v8f_zero(), v8f_zero(), v8f_zero()};
  float mrun[8], lrun[8];
#pragma unroll
  for (int j = 0; j < 8; ++j) { mrun[j] = -1e30f; lrun[j] = 0.f; }

  for (int k0 = 0; k0 <= q0 + 15; k0 += 32) {
    // ---- scores: 16 queries x 32 keys = two 16x16 C tiles
    v8f S0 = v8f_zero();
    v8f S1 = v8f_zero();
#pragma unroll
    for (int c = 0; c < 4; ++c) {
      ABFrag bk;
      const uint4* bp0 =
          (const uint4*)(kh + (size_t)(k0 + mrow) * HDIM + c * 32 + hi * 8);
      bk.u[0] = bp0[0];
      bk.u[1] = bp0[2];
      S0 = wmma_bf16(Aq[c].v, bk.v, S0);
      const uint4* bp1 =
          (const uint4*)(kh + (size_t)(k0 + 16 + mrow) * HDIM + c * 32 + hi * 8);
      bk.u[0] = bp1[0];
      bk.u[1] = bp1[2];
      S1 = wmma_bf16(Aq[c].v, bk.v, S1);
    }
    // ---- causal mask + online softmax (rows live in 16-lane halves)
#pragma unroll
    for (int j = 0; j < 8; ++j) {
      const int qidx = q0 + j + hi * 8;
      float s0 = (k0 + mrow      <= qidx) ? S0[j] : -1e30f;
      float s1 = (k0 + 16 + mrow <= qidx) ? S1[j] : -1e30f;
      float rm = fmaxf(s0, s1);
      rm = fmaxf(rm, __shfl_xor(rm, 1, 32));
      rm = fmaxf(rm, __shfl_xor(rm, 2, 32));
      rm = fmaxf(rm, __shfl_xor(rm, 4, 32));
      rm = fmaxf(rm, __shfl_xor(rm, 8, 32));
      const float mnew  = fmaxf(mrun[j], rm);
      const float alpha = __expf(mrun[j] - mnew);
      const float p0 = __expf(s0 - mnew);
      const float p1 = __expf(s1 - mnew);
      float rs = p0 + p1;
      rs += __shfl_xor(rs, 1, 32);
      rs += __shfl_xor(rs, 2, 32);
      rs += __shfl_xor(rs, 4, 32);
      rs += __shfl_xor(rs, 8, 32);
      lrun[j] = lrun[j] * alpha + rs;
      mrun[j] = mnew;
#pragma unroll
      for (int dt = 0; dt < 8; ++dt) O[dt][j] *= alpha;
      const int r = j + hi * 8;
      P[r * 32 + mrow]      = p0;
      P[r * 32 + 16 + mrow] = p1;
    }
    asm volatile("s_wait_dscnt 0" ::: "memory");   // per-wave LDS transpose fence
    // ---- read P back in A-operand layout, convert to bf16
    ABFrag pa;
    {
      const float* pr = P + mrow * 32;
      float4 f0 = *(const float4*)(pr + hi * 8);
      float4 f1 = *(const float4*)(pr + hi * 8 + 4);
      float4 f2 = *(const float4*)(pr + 16 + hi * 8);
      float4 f3 = *(const float4*)(pr + 16 + hi * 8 + 4);
      pa.h[0]  = f2bf(f0.x); pa.h[1]  = f2bf(f0.y);
      pa.h[2]  = f2bf(f0.z); pa.h[3]  = f2bf(f0.w);
      pa.h[4]  = f2bf(f1.x); pa.h[5]  = f2bf(f1.y);
      pa.h[6]  = f2bf(f1.z); pa.h[7]  = f2bf(f1.w);
      pa.h[8]  = f2bf(f2.x); pa.h[9]  = f2bf(f2.y);
      pa.h[10] = f2bf(f2.z); pa.h[11] = f2bf(f2.w);
      pa.h[12] = f2bf(f3.x); pa.h[13] = f2bf(f3.y);
      pa.h[14] = f2bf(f3.z); pa.h[15] = f2bf(f3.w);
    }
    // ---- O += P @ V  (V transposed: contiguous key runs per dim-lane)
#pragma unroll
    for (int dt = 0; dt < 8; ++dt) {
      ABFrag bv;
      const uint4* bp =
          (const uint4*)(vh + (size_t)(dt * 16 + mrow) * SEQ + k0 + hi * 8);
      bv.u[0] = bp[0];
      bv.u[1] = bp[2];
      O[dt] = wmma_bf16(pa.v, bv.v, O[dt]);
    }
  }

  // ---- finalize: O / l, store y bf16 [B][S][H*128]
  const int batch = bh >> 4;
  const int h = bh & 15;
#pragma unroll
  for (int j = 0; j < 8; ++j) {
    const float inv = 1.0f / lrun[j];
    const int srow = q0 + j + hi * 8;
    const size_t base = ((size_t)batch * SEQ + srow) * DIN + h * HDIM;
#pragma unroll
    for (int dt = 0; dt < 8; ++dt) {
      yb[base + dt * 16 + mrow] = f2bf(O[dt][j] * inv);
    }
  }
}

// ---------------------------------------------------------------------------
// Host-side orchestration
// ---------------------------------------------------------------------------
extern "C" void kernel_launch(void* const* d_in, const int* in_sizes, int n_in,
                              void* d_out, int out_size, void* d_ws, size_t ws_size,
                              hipStream_t stream) {
  const float* x     = (const float*)d_in[0];   // [2][2048][2048]
  const float* Wqkv  = (const float*)d_in[1];   // [2048][6144]
  const float* Wproj = (const float*)d_in[2];   // [2048][2048]
  float* out = (float*)d_out;                   // [2][2048][2048]

  char* ws = (char*)d_ws;
  size_t off = 0;
  unsigned short* xb     = (unsigned short*)(ws + off); off += (size_t)MTOT * DIN * 2;
  unsigned short* wqkvT  = (unsigned short*)(ws + off); off += (size_t)NQKV * DIN * 2;
  unsigned short* wprojT = (unsigned short*)(ws + off); off += (size_t)DIN * DIN * 2;
  unsigned short* q_bf   = (unsigned short*)(ws + off); off += (size_t)MTOT * DIN * 2;
  unsigned short* k_bf   = (unsigned short*)(ws + off); off += (size_t)MTOT * DIN * 2;
  unsigned short* vT_bf  = (unsigned short*)(ws + off); off += (size_t)MTOT * DIN * 2;
  unsigned short* y_bf   = (unsigned short*)(ws + off); off += (size_t)MTOT * DIN * 2;

  // 1) convert x to bf16
  {
    int n = MTOT * DIN;
    cvt_bf16_kernel<<<n / 256, 256, 0, stream>>>(x, xb, n);
  }
  // 2) transpose+convert weights
  {
    long long n = (long long)DIN * NQKV;
    transpose_bf16_kernel<<<(unsigned)(n / 256), 256, 0, stream>>>(Wqkv, wqkvT,
                                                                   DIN, NQKV);
  }
  {
    long long n = (long long)DIN * DIN;
    transpose_bf16_kernel<<<(unsigned)(n / 256), 256, 0, stream>>>(Wproj, wprojT,
                                                                   DIN, DIN);
  }
  // 3) QKV GEMM + RoPE
  gemm_qkv_rope_kernel<<<dim3(NQKV / 128, MTOT / 128), 256, 0, stream>>>(
      xb, wqkvT, q_bf, k_bf, vT_bf);
  // 4) flash attention
  attn_fwd_kernel<<<dim3(SEQ / 64, 2 * NHEAD), 128, 0, stream>>>(q_bf, k_bf,
                                                                 vT_bf, y_bf);
  // 5) output projection
  gemm_proj_kernel<<<dim3(DIN / 128, MTOT / 128), 256, 0, stream>>>(y_bf, wprojT,
                                                                    out);
  (void)in_sizes; (void)n_in; (void)out_size; (void)ws_size;
}
